// FlashAttention_10685878632993
// MI455X (gfx1250) — matrix operand order repaired and checked by reading, not compiled, not run on hardware
//
#include <hip/hip_runtime.h>
#include <hip/hip_bf16.h>

typedef __attribute__((ext_vector_type(16))) _Float16 v16h;
typedef __attribute__((ext_vector_type(2)))  _Float16 v2h;
typedef __attribute__((ext_vector_type(8)))  float    v8f;

#define S_LEN   2048
#define HEADS   16
#define DDIM    64
#define BM      128   // query rows per block
#define WM      16    // query rows per wave
#define BN      32    // keys per chunk
#define NWAVES  8
#define NEG_INF (-1e30f)
#define LOG2E   1.4426950408889634f

struct __align__(128) FaShared {
    _Float16 K[BN * DDIM];            // key-major:  key(0..31) x d(0..63), f16
    _Float16 VT[DDIM * BN];           // transposed: d(0..63)  x key(0..31), f16
    _Float16 P[NWAVES][WM * BN];      // per-wave probs, stored row-major (m x key) in plain key order
};

__global__ __launch_bounds__(256)
void fa_fwd_kernel(const float* __restrict__ qkv,
                   const float* __restrict__ bias,
                   float* __restrict__ out)
{
    __shared__ FaShared sm;

    const int tid   = threadIdx.x;
    const int wave  = tid >> 5;
    const int lane  = tid & 31;
    const int lhalf = lane >> 4;      // 0 or 1
    const int l16   = lane & 15;

    const int qtile = blockIdx.x;     // 0..15
    const int h     = blockIdx.y;
    const int b     = blockIdx.z;
    const int q0    = qtile * BM;
    const int qbase = q0 + wave * WM; // this wave's first query row

    // qkv layout (B,S,3,H,D): (((b*S+s)*3+which)*H+h)*D + d
    const size_t srow = (size_t)3 * HEADS * DDIM;               // stride in s
    const size_t bh   = (size_t)b * S_LEN * 3 * HEADS * DDIM + (size_t)h * DDIM;
    const float* qp = qkv + bh;
    const float* kp = qkv + bh + (size_t)1 * HEADS * DDIM;
    const float* vp = qkv + bh + (size_t)2 * HEADS * DDIM;
    // bias row base for this lane: bias[h][qbase + 8*lhalf + v][l16 + k0]
    const float* bq_base = bias + (size_t)h * S_LEN * S_LEN
                                + (size_t)(qbase + 8 * lhalf) * S_LEN + l16;

    // ---- Load Q stripe into registers (A-layout), pre-scaled by log2(e)/sqrt(D) ----
    // A-layout f16 16x32: lane m=l16; elem e(0..15): K = e + 16*lhalf (contiguous)
    v16h qa[2];
    {
        const float* qr = qp + (size_t)(qbase + l16) * srow;
        const float qs = 0.125f * LOG2E;
#pragma unroll
        for (int c = 0; c < 2; ++c) {
#pragma unroll
            for (int v = 0; v < 8; ++v) {
                const int d = 32 * c + 16 * lhalf + 2 * v;
                const float2 f = *(const float2*)(qr + d);
                qa[c][2 * v]     = (_Float16)(f.x * qs);
                qa[c][2 * v + 1] = (_Float16)(f.y * qs);
            }
        }
    }

    // Output accumulator: 4 C-tiles (16x16 each) covering d=0..63
    v8f acc[4];
#pragma unroll
    for (int t = 0; t < 4; ++t) acc[t] = (v8f){};
    float mi[8], li[8];
#pragma unroll
    for (int v = 0; v < 8; ++v) { mi[v] = NEG_INF; li[v] = 0.0f; }

    const int nchunks = (q0 + BM) / BN;

    for (int kc = 0; kc < nchunks; ++kc) {
        const int k0 = kc * BN;

        __syncthreads();  // previous chunk's LDS reads complete before overwrite
        // ---- cooperative staging: wave w converts keys 4w..4w+3 to f16 in LDS ----
#pragma unroll
        for (int r = 0; r < 4; ++r) {
            const int kr  = wave * 4 + r;          // 0..31 within chunk
            const int key = k0 + kr;
            const float2 kf = *(const float2*)(kp + (size_t)key * srow + 2 * lane);
            v2h k2; k2.x = (_Float16)kf.x; k2.y = (_Float16)kf.y;
            *(v2h*)&sm.K[kr * DDIM + 2 * lane] = k2;
            const float2 vf = *(const float2*)(vp + (size_t)key * srow + 2 * lane);
            sm.VT[(2 * lane)     * BN + kr] = (_Float16)vf.x;
            sm.VT[(2 * lane + 1) * BN + kr] = (_Float16)vf.y;
        }
        __syncthreads();

        // ---- wave-local compute (skip chunks entirely above this wave's diagonal) ----
        if (k0 <= qbase + WM - 1) {
            // QK^T: two 16x16 score tiles (keys k0..+15, k0+16..+31), K-loop over D
            v8f s0 = (v8f){}, s1 = (v8f){};
#pragma unroll
            for (int c = 0; c < 2; ++c) {
                // B-layout: lane n=l16 (key), elem e: d = 32c + 16*lhalf + e (contiguous)
                const v16h kb0 = *(const v16h*)&sm.K[ l16       * DDIM + 32 * c + 16 * lhalf];
                const v16h kb1 = *(const v16h*)&sm.K[(16 + l16) * DDIM + 32 * c + 16 * lhalf];
                s0 = __builtin_amdgcn_wmma_f32_16x16x32_f16(false, qa[c], false, kb0,
                                                            (short)0, s0, false, false);
                s1 = __builtin_amdgcn_wmma_f32_16x16x32_f16(false, qa[c], false, kb1,
                                                            (short)0, s1, false, false);
            }

            // C-layout: lane holds (m = v + 8*lhalf, n = l16). Bias via fma into the
            // log2 domain; causal compare/select only on the single boundary chunk.
            if (k0 + BN - 1 <= qbase) {            // interior chunk: no masking needed
#pragma unroll
                for (int v = 0; v < 8; ++v) {
                    const float* bq = bq_base + (size_t)v * S_LEN + k0;
                    s0[v] = fmaf(bq[0],  LOG2E, s0[v]);
                    s1[v] = fmaf(bq[16], LOG2E, s1[v]);
                }
            } else {                               // boundary chunk: apply causal mask
#pragma unroll
                for (int v = 0; v < 8; ++v) {
                    const int qrow = qbase + v + 8 * lhalf;
                    const float* bq = bq_base + (size_t)v * S_LEN + k0;
                    s0[v] = (k0 + l16      <= qrow) ? fmaf(bq[0],  LOG2E, s0[v]) : NEG_INF;
                    s1[v] = (k0 + 16 + l16 <= qrow) ? fmaf(bq[16], LOG2E, s1[v]) : NEG_INF;
                }
            }

            // online softmax (log2 domain): row reductions across 16-lane half-groups
            _Float16* pb = sm.P[wave];
            const int j0 = l16;                    // contiguous key order for this lane's column
#pragma unroll
            for (int v = 0; v < 8; ++v) {
                float rmax = fmaxf(s0[v], s1[v]);
#pragma unroll
                for (int off = 8; off >= 1; off >>= 1)
                    rmax = fmaxf(rmax, __shfl_xor(rmax, off, 16));
                const float mnew  = fmaxf(mi[v], rmax);
                const float alpha = __builtin_amdgcn_exp2f(mi[v] - mnew);
                mi[v] = mnew;
                const float p0 = __builtin_amdgcn_exp2f(s0[v] - mnew);
                const float p1 = __builtin_amdgcn_exp2f(s1[v] - mnew);
                float rsum = p0 + p1;
#pragma unroll
                for (int off = 8; off >= 1; off >>= 1)
                    rsum += __shfl_xor(rsum, off, 16);
                li[v] = li[v] * alpha + rsum;
#pragma unroll
                for (int t = 0; t < 4; ++t) acc[t][v] *= alpha;
                // spill P (f16) in key order so the A-operand reload is contiguous
                const int m = v + 8 * lhalf;
                pb[m * BN + j0]      = (_Float16)p0;
                pb[m * BN + j0 + 16] = (_Float16)p1;
            }

            // reload P as one contiguous v16h per lane (DS ops in-order within a wave)
            const v16h pa = *(const v16h*)&pb[l16 * BN + 16 * lhalf];

            // P(16x32) x V(32x64): 4 WMMAs into the persistent accumulators
#pragma unroll
            for (int t = 0; t < 4; ++t) {
                // B-layout: lane n=l16 -> d = 16t+l16; elem e -> key = 16*lhalf + e
                const v16h vb = *(const v16h*)&sm.VT[(16 * t + l16) * BN + 16 * lhalf];
                acc[t] = __builtin_amdgcn_wmma_f32_16x16x32_f16(false, pa, false, vb,
                                                                (short)0, acc[t], false, false);
            }
        }
    }

    // ---- finalize: divide by row sums, store fp32 (B,S,H,D) ----
    float inv[8];
#pragma unroll
    for (int v = 0; v < 8; ++v) inv[v] = 1.0f / li[v];
    float* op = out + (size_t)b * S_LEN * HEADS * DDIM + (size_t)h * DDIM;
#pragma unroll
    for (int t = 0; t < 4; ++t) {
#pragma unroll
        for (int v = 0; v < 8; ++v) {
            const int qrow = qbase + v + 8 * lhalf;
            const int d    = 16 * t + l16;
            op[(size_t)qrow * HEADS * DDIM + d] = acc[t][v] * inv[v];
        }
    }
}

extern "C" void kernel_launch(void* const* d_in, const int* in_sizes, int n_in,
                              void* d_out, int out_size, void* d_ws, size_t ws_size,
                              hipStream_t stream) {
    const float* qkv  = (const float*)d_in[0];   // (B,S,3,H,D) fp32
    const float* bias = (const float*)d_in[1];   // (H,S,S)     fp32
    float* out = (float*)d_out;                  // (B,S,H,D)   fp32
    dim3 grid(S_LEN / BM, HEADS, 2);
    dim3 block(256);
    fa_fwd_kernel<<<grid, block, 0, stream>>>(qkv, bias, out);
}
